// ShiftedWindowAttention_88381837017290
// MI455X (gfx1250) — matrix operand order repaired and checked by reading, not compiled, not run on hardware
//
#include <hip/hip_runtime.h>

// ---------------------------------------------------------------------------
// Shifted-window attention (Swin), fully fused per-window kernel for gfx1250.
// B=32 H=W=56 C=384 ws=7 shift=3 heads=12 hd=32 ; N=49 tokens padded to 64.
// All three GEMMs run on v_wmma_f32_16x16x32_bf16 (wave32 WMMA).
// x tile staged via GLOBAL_LOAD_ASYNC_TO_LDS_B128 (ASYNCcnt-tracked).
// QKV GEMM: dual N-strip per wave -> 8 WMMAs per k-step per 4 LDS A-loads.
// ---------------------------------------------------------------------------

typedef __attribute__((ext_vector_type(16))) __bf16        v16bf;
typedef __attribute__((ext_vector_type(8)))  float         v8f;
typedef __attribute__((ext_vector_type(4)))  unsigned int  u32x4;
typedef __attribute__((ext_vector_type(4)))  float         f32x4;

#define B_N    32
#define H_N    56
#define W_N    56
#define C_N    384
#define C3_N   1152
#define SS_N   3
#define NH_N   12
#define HD_N   32
#define NTOK   49
#define NPAD   64
#define NWIN   2048
#define SCALE_Q 0.17677669529663687f   // 1/sqrt(32)

union BF16Frag { v16bf v; u32x4 q[2]; unsigned short h[16]; };
union AccFrag  { v8f   v; float f[8]; };

__device__ __forceinline__ unsigned short f2bf(float x) {
  unsigned u = __float_as_uint(x);
  u += 0x7FFFu + ((u >> 16) & 1u);      // round-to-nearest-even
  return (unsigned short)(u >> 16);
}

// A-matrix fragment (16x32 bf16). CDNA5 layout: row M = lane%16; lanes 0-15
// hold K = {0..7,16..23}, lanes 16-31 hold K = {8..15,24..31}; 2 halves/VGPR.
__device__ __forceinline__ v16bf load_frag_A(const unsigned short* base, int stride,
                                             int row, int kbase, int lane) {
  const unsigned short* p = base + row * stride + kbase + ((lane & 16) ? 8 : 0);
  BF16Frag f;
  f.q[0] = *(const u32x4*)(p);        // K +0..7
  f.q[1] = *(const u32x4*)(p + 16);   // K +16..23
  return f.v;
}

// B-matrix fragment (32x16 bf16) from an N-major (transposed) buffer:
// column N = lane%16; lanes 0-15 hold K={0..7,16..23}, lanes 16-31 hold K={8..15,24..31}.
__device__ __forceinline__ v16bf load_frag_B(const unsigned short* base, int stride,
                                             int nBase, int kbase, int lane) {
  const unsigned short* p = base + (nBase + (lane & 15)) * stride + kbase + ((lane & 16) ? 8 : 0);
  BF16Frag f;
  f.q[0] = *(const u32x4*)(p);
  f.q[1] = *(const u32x4*)(p + 16);
  return f.v;
}

__device__ __forceinline__ v8f wmma_bf16(v16bf a, v16bf b, v8f c) {
  return __builtin_amdgcn_wmma_f32_16x16x32_bf16(false, a, false, b, (short)0, c, false, false);
}

// ---------------------------------------------------------------------------
// Pre-pass: convert weights f32 -> bf16 and transpose to N-major so WMMA
// B-fragments are two k-contiguous 16-byte per-lane reads.
// ---------------------------------------------------------------------------
__global__ void swin_wconvert_88381837017290(const float* __restrict__ qkv_w,
                                             const float* __restrict__ proj_w,
                                             unsigned short* __restrict__ qkvT,
                                             unsigned short* __restrict__ projT) {
  int idx = blockIdx.x * blockDim.x + threadIdx.x;
  if (idx < C3_N * C_N) {               // qkvT[n][k] = qkv_w[k][n]
    int nn = idx / C_N, kk = idx % C_N;
    qkvT[idx] = f2bf(qkv_w[kk * C3_N + nn]);
  }
  if (idx < C_N * C_N) {                // projT[n][k] = proj_w[k][n]
    int nn = idx / C_N, kk = idx % C_N;
    projT[idx] = f2bf(proj_w[kk * C_N + nn]);
  }
}

// ---------------------------------------------------------------------------
// Fused per-window kernel. Grid = 2048 windows, block = 256 threads (8 waves).
// Dynamic LDS layout (halves unless noted):
//   Xs   [64][384]   input tile bf16, reused later as attention-output tile
//   QKVs [64][1152]  Q|K|V bf16 (phase-0: f32 async-staging area for x tile)
//   Vt   [384][64]   V transposed (dim-major) for PV B-fragments
//   Pb   8 x [16][64] per-wave softmax(P) staging for A-fragment relayout
//   biasLds[169*12]  relative-position-bias table (f32)
// Total = 262144 + 8112 = 270256 bytes (< 320 KB / WGP).
// ---------------------------------------------------------------------------
__global__ void swin_attn_fused_88381837017290(const float* __restrict__ x,
                                               const unsigned short* __restrict__ qkvT,
                                               const float* __restrict__ qkv_b,
                                               const unsigned short* __restrict__ projT,
                                               const float* __restrict__ proj_b,
                                               const float* __restrict__ rpb,
                                               float* __restrict__ out) {
  extern __shared__ unsigned short smem[];
  unsigned short* Xs   = smem;                       // 64*384
  unsigned short* QKVs = Xs + NPAD * C_N;            // 64*1152
  unsigned short* Vt   = QKVs + NPAD * C3_N;         // 384*64
  unsigned short* Pb   = Vt + C_N * NPAD;            // 8*16*64
  float* biasLds = (float*)(Pb + 8 * 16 * 64);       // 169*12 f32

  const int tid  = threadIdx.x;
  const int lane = tid & 31;
  const int wave = tid >> 5;
  const int c0   = lane & 15;
  const int hh   = (lane & 16) ? 8 : 0;              // D-layout row offset

  const int win = blockIdx.x;
  const int b   = win >> 6;
  const int wr  = win & 63;
  const int wi  = wr >> 3;
  const int wj  = wr & 7;

  // ---- Phase 0: async-stage shifted window tile into LDS (f32), convert ---
  float* stageF = (float*)QKVs;                      // 75,264 B staging, dead until phase 1
  for (int idx = tid; idx < NTOK * (C_N / 4); idx += 256) {
    int t  = idx / (C_N / 4);
    int c4 = idx % (C_N / 4);
    int ti = t / 7, tj = t % 7;
    int sr = (wi * 7 + ti + SS_N) % H_N;             // roll(-shift) gather
    int sc = (wj * 7 + tj + SS_N) % W_N;
    const float* gp = x + ((size_t)(b * H_N + sr) * W_N + sc) * C_N + c4 * 4;
    unsigned ldsOff = (unsigned)(size_t)(stageF + (size_t)idx * 4);  // LDS aperture: addr[31:0]
    asm volatile("global_load_async_to_lds_b128 %0, %1, off"
                 :: "v"(ldsOff), "v"((unsigned long long)(size_t)gp)
                 : "memory");
  }
  for (int idx = tid; idx < (NPAD - NTOK) * (C_N / 4); idx += 256) {
    int t  = NTOK + idx / (C_N / 4);
    int c4 = idx % (C_N / 4);
    unsigned* dst = (unsigned*)(Xs + t * C_N + c4 * 4);
    dst[0] = 0u; dst[1] = 0u;
  }
  for (int idx = tid; idx < 169 * NH_N; idx += 256) biasLds[idx] = rpb[idx];
  asm volatile("s_wait_asynccnt 0" ::: "memory");    // our async copies landed in LDS
  for (int idx = tid; idx < NTOK * (C_N / 4); idx += 256) {
    int t  = idx / (C_N / 4);
    int c4 = idx % (C_N / 4);
    f32x4 v = *(const f32x4*)(stageF + (size_t)idx * 4);  // own chunks only
    unsigned lo = (unsigned)f2bf(v[0]) | ((unsigned)f2bf(v[1]) << 16);
    unsigned hi = (unsigned)f2bf(v[2]) | ((unsigned)f2bf(v[3]) << 16);
    unsigned* dst = (unsigned*)(Xs + t * C_N + c4 * 4);
    dst[0] = lo; dst[1] = hi;
  }
  __syncthreads();

  // ---- Phase 1: QKV GEMM  (64x384)x(384x1152) on WMMA bf16 ----------------
  // Dual N-strip per wave: per k-step, 2 global B-frags + 4 LDS A-frags feed
  // 8 WMMAs (every A reused twice back-to-back, halving LDS-wait density).
  for (int p = wave; p < 36; p += 8) {
    int nB0 = p * 16;
    int nB1 = (p + 36) * 16;
    if (p + 8 < 36) {                                    // stream next weight strips
      __builtin_prefetch(qkvT + (size_t)(nB0 + 128) * C_N, 0, 1);
      __builtin_prefetch(qkvT + (size_t)(nB1 + 128) * C_N, 0, 1);
    }
    AccFrag acc0[4] = {}, acc1[4] = {};
#pragma unroll
    for (int k = 0; k < 12; ++k) {
      v16bf b0 = load_frag_B(qkvT, C_N, nB0, k * 32, lane);
      v16bf b1 = load_frag_B(qkvT, C_N, nB1, k * 32, lane);
#pragma unroll
      for (int m = 0; m < 4; ++m) {
        v16bf a = load_frag_A(Xs, C_N, m * 16 + c0, k * 32, lane);
        acc0[m].v = wmma_bf16(a, b0, acc0[m].v);
        acc1[m].v = wmma_bf16(a, b1, acc1[m].v);
      }
    }
#pragma unroll
    for (int s = 0; s < 2; ++s) {
      AccFrag* acc = s ? acc1 : acc0;
      int   n     = (s ? nB1 : nB0) + c0;
      float bias  = qkv_b[n];
      float scale = (n < C_N) ? SCALE_Q : 1.0f;          // pre-scale Q
      bool  isV   = (n >= 2 * C_N);
#pragma unroll
      for (int m = 0; m < 4; ++m)
#pragma unroll
        for (int i = 0; i < 8; ++i) {
          int row = m * 16 + i + hh;
          unsigned short bv = f2bf((acc[m].f[i] + bias) * scale);
          QKVs[row * C3_N + n] = bv;
          if (isV) Vt[(n - 2 * C_N) * NPAD + row] = bv;  // V transposed copy
        }
    }
  }
  __syncthreads();

  // ---- Phase 2: per-head attention  S=QK^T, softmax(+rpb), O=PV -----------
  unsigned short* Pw = Pb + wave * (16 * 64);
  for (int h = wave; h < NH_N; h += 8) {
    int qOff = h * HD_N;
    int kOff = C_N + h * HD_N;
    v16bf Kf[4];
#pragma unroll
    for (int n = 0; n < 4; ++n) Kf[n] = load_frag_B(QKVs, C3_N, n * 16, kOff, lane);
    v16bf Vf[2][2];
#pragma unroll
    for (int kc = 0; kc < 2; ++kc)
#pragma unroll
      for (int dt = 0; dt < 2; ++dt)
        Vf[kc][dt] = load_frag_B(Vt, NPAD, h * HD_N + dt * 16, kc * 32, lane);

    for (int m = 0; m < 4; ++m) {
      v16bf qf = load_frag_A(QKVs, C3_N, m * 16 + c0, qOff, lane);
      AccFrag S[4] = {};
#pragma unroll
      for (int n = 0; n < 4; ++n) S[n].v = wmma_bf16(qf, Kf[n], S[n].v);

      // masked softmax over 64 key slots (49 valid), rows live in half-waves
#pragma unroll
      for (int i = 0; i < 8; ++i) {
        int r = m * 16 + i + hh;
        float sv[4];
#pragma unroll
        for (int n = 0; n < 4; ++n) {
          int c = n * 16 + c0;
          float s = S[n].f[i];
          if (r < NTOK && c < NTOK) {
            int dy = r / 7 - c / 7 + 6;
            int dx = r % 7 - c % 7 + 6;
            s += biasLds[(dy * 13 + dx) * NH_N + h];
          } else {
            s = -1e30f;
          }
          sv[n] = s;
        }
        float mx = fmaxf(fmaxf(sv[0], sv[1]), fmaxf(sv[2], sv[3]));
#pragma unroll
        for (int off = 1; off < 16; off <<= 1) mx = fmaxf(mx, __shfl_xor(mx, off, 16));
        float e[4], sum = 0.f;
#pragma unroll
        for (int n = 0; n < 4; ++n) { e[n] = __expf(sv[n] - mx); sum += e[n]; }
#pragma unroll
        for (int off = 1; off < 16; off <<= 1) sum += __shfl_xor(sum, off, 16);
        float inv = 1.0f / fmaxf(sum, 1e-20f);
        int rl = i + hh;
#pragma unroll
        for (int n = 0; n < 4; ++n) Pw[rl * 64 + n * 16 + c0] = f2bf(e[n] * inv);
      }

      AccFrag O[2] = {};
#pragma unroll
      for (int kc = 0; kc < 2; ++kc) {
        v16bf pf = load_frag_A(Pw, 64, c0, kc * 32, lane);
#pragma unroll
        for (int dt = 0; dt < 2; ++dt) O[dt].v = wmma_bf16(pf, Vf[kc][dt], O[dt].v);
      }
#pragma unroll
      for (int dt = 0; dt < 2; ++dt)
#pragma unroll
        for (int i = 0; i < 8; ++i) {
          int row = m * 16 + i + hh;
          Xs[row * C_N + qOff + dt * 16 + c0] = f2bf(O[dt].f[i]);  // Xs reused as O-tile
        }
    }
  }
  __syncthreads();

  // ---- Phase 3: output projection  (64x384)x(384x384), scatter f32 --------
  for (int nt = wave; nt < 24; nt += 8) {
    int nBase = nt * 16;
    AccFrag acc[4] = {};
#pragma unroll
    for (int k = 0; k < 12; ++k) {
      v16bf bf = load_frag_B(projT, C_N, nBase, k * 32, lane);
#pragma unroll
      for (int m = 0; m < 4; ++m) {
        v16bf a = load_frag_A(Xs, C_N, m * 16 + c0, k * 32, lane);
        acc[m].v = wmma_bf16(a, bf, acc[m].v);
      }
    }
    float bias = proj_b[nBase + c0];
#pragma unroll
    for (int m = 0; m < 4; ++m)
#pragma unroll
      for (int i = 0; i < 8; ++i) {
        int t = m * 16 + i + hh;
        if (t < NTOK) {
          int ti = t / 7, tj = t % 7;
          int dr = (wi * 7 + ti + SS_N) % H_N;           // roll(+shift) scatter
          int dc = (wj * 7 + tj + SS_N) % W_N;
          out[((size_t)(b * H_N + dr) * W_N + dc) * C_N + nBase + c0] = acc[m].f[i] + bias;
        }
      }
  }
}

// ---------------------------------------------------------------------------
extern "C" void kernel_launch(void* const* d_in, const int* in_sizes, int n_in,
                              void* d_out, int out_size, void* d_ws, size_t ws_size,
                              hipStream_t stream) {
  (void)in_sizes; (void)n_in; (void)out_size; (void)ws_size;
  const float* x      = (const float*)d_in[0];
  const float* qkv_w  = (const float*)d_in[1];
  const float* qkv_b  = (const float*)d_in[2];
  const float* proj_w = (const float*)d_in[3];
  const float* proj_b = (const float*)d_in[4];
  const float* rpb    = (const float*)d_in[5];
  float* out = (float*)d_out;

  // Workspace: bf16 transposed weights (1.18 MB total)
  unsigned short* qkvT  = (unsigned short*)d_ws;
  unsigned short* projT = qkvT + (size_t)C3_N * C_N;

  int convBlocks = (C3_N * C_N + 255) / 256;
  swin_wconvert_88381837017290<<<convBlocks, 256, 0, stream>>>(qkv_w, proj_w, qkvT, projT);

  size_t smemBytes = (size_t)(NPAD * C_N + NPAD * C3_N + C_N * NPAD + 8 * 16 * 64) * 2
                   + (size_t)169 * NH_N * 4;  // 270,256 B
  (void)hipFuncSetAttribute((const void*)swin_attn_fused_88381837017290,
                            hipFuncAttributeMaxDynamicSharedMemorySize, (int)smemBytes);
  swin_attn_fused_88381837017290<<<NWIN, 256, smemBytes, stream>>>(
      x, qkvT, qkv_b, projT, proj_b, rpb, out);
}
